// SNNTokenizer_37134287241278
// MI455X (gfx1250) — matrix operand order repaired and checked by reading, not compiled, not run on hardware
//
#include <hip/hip_runtime.h>

typedef _Float16 h16;
typedef __attribute__((ext_vector_type(16))) _Float16 v16h;
typedef __attribute__((ext_vector_type(8)))  _Float16 v8h;
typedef __attribute__((ext_vector_type(8)))  float    v8f;

#define B_  256
#define T_  512
#define DC_ 128
#define DH_ 256
#define DO_ 768
#define M_  (B_*T_)   /* 131072 rows */

// ---- helpers -------------------------------------------------------------

static __device__ inline v8f wmma_f16(v16h a, v16h b, v8f c) {
  // D = A(16x32 f16) * B(32x16 f16) + C(16x16 f32)
  return __builtin_amdgcn_wmma_f32_16x16x32_f16(false, a, false, b, (short)0, c,
                                                false, false);
}

// Build A fragment from a row-major f16 row (lane covers M=lane&15).
static __device__ inline v16h load_a_h16(const h16* __restrict__ arow,
                                         int k0, int hi) {
  v8h lo = *(const v8h*)(arow + k0 + hi * 8);
  v8h hh = *(const v8h*)(arow + k0 + 16 + hi * 8);
  v16h a;
#pragma unroll
  for (int e = 0; e < 8; ++e) { a[e] = lo[e]; a[e + 8] = hh[e]; }
  return a;
}

// ---- f32 -> f16 weight pre-conversion (vectorized, n % 4 == 0) -----------
__global__ void __launch_bounds__(256) k_tohalf(const float* __restrict__ src,
                                                h16* __restrict__ dst, int n4) {
  const int i = blockIdx.x * 256 + threadIdx.x;
  if (i < n4) {
    float4 f = *(const float4*)(src + i * 4);
    h16* d = dst + i * 4;
    d[0] = (h16)f.x; d[1] = (h16)f.y; d[2] = (h16)f.z; d[3] = (h16)f.w;
  }
}

// ---- GEMM1: gather(embH[ids]) @ W1H + b1 -> Y (f32) ----------------------
__global__ void __launch_bounds__(256) k_gemm1(const int* __restrict__ ids,
                                               const h16* __restrict__ embH,
                                               const h16* __restrict__ W1H,
                                               const float* __restrict__ b1,
                                               float* __restrict__ Y) {
  const int lane = threadIdx.x & 31, wave = threadIdx.x >> 5;
  const int r = lane & 15, hi = lane >> 4;
  const int m0 = blockIdx.x * 16;
  const int n0 = (blockIdx.y * 8 + wave) * 16;
  const int row = ids[m0 + r];                       // fused embedding gather
  const h16* arow = embH + (size_t)row * DC_;
  v8f acc;
#pragma unroll
  for (int e = 0; e < 8; ++e) acc[e] = 0.0f;
#pragma unroll
  for (int k0 = 0; k0 < DC_; k0 += 32) {
    v16h a = load_a_h16(arow, k0, hi);
    v16h b;
#pragma unroll
    for (int e = 0; e < 16; ++e)
      b[e] = W1H[(size_t)(k0 + (e >> 3) * 16 + hi * 8 + (e & 7)) * DH_ + n0 + r];
    acc = wmma_f16(a, b, acc);
  }
  const float bias = b1[n0 + r];
#pragma unroll
  for (int v = 0; v < 8; ++v)
    Y[(size_t)(m0 + v + hi * 8) * DH_ + n0 + r] = acc[v] + bias;
}

// ---- GEMM2: S1(f16) @ W2H + b2 -> Y (f32) --------------------------------
__global__ void __launch_bounds__(256) k_gemm2(const h16* __restrict__ A,
                                               const h16* __restrict__ W2H,
                                               const float* __restrict__ b2,
                                               float* __restrict__ Y) {
  const int lane = threadIdx.x & 31, wave = threadIdx.x >> 5;
  const int r = lane & 15, hi = lane >> 4;
  const int m0 = blockIdx.x * 16;
  const int n0 = (blockIdx.y * 8 + wave) * 16;
  const h16* arow = A + (size_t)(m0 + r) * DH_;
  v8f acc;
#pragma unroll
  for (int e = 0; e < 8; ++e) acc[e] = 0.0f;
#pragma unroll
  for (int k0 = 0; k0 < DH_; k0 += 32) {
    v16h a = load_a_h16(arow, k0, hi);
    v16h b;
#pragma unroll
    for (int e = 0; e < 16; ++e)
      b[e] = W2H[(size_t)(k0 + (e >> 3) * 16 + hi * 8 + (e & 7)) * DH_ + n0 + r];
    acc = wmma_f16(a, b, acc);
  }
  const float bias = b2[n0 + r];
#pragma unroll
  for (int v = 0; v < 8; ++v)
    Y[(size_t)(m0 + v + hi * 8) * DH_ + n0 + r] = acc[v] + bias;
}

// ---- BN stats: deterministic two-pass (no float atomics) -----------------
__global__ void __launch_bounds__(256) k_bnstats(const float* __restrict__ Y,
                                                 float* __restrict__ part) {
  const int d = threadIdx.x;                     // channel
  const size_t base = (size_t)blockIdx.x * 512;  // 256 blocks x 512 rows
  float s = 0.f, q = 0.f;
  for (int i = 0; i < 512; ++i) {
    float x = Y[(base + i) * DH_ + d];
    s += x; q += x * x;
  }
  part[(size_t)blockIdx.x * 512 + d]       = s;
  part[(size_t)blockIdx.x * 512 + 256 + d] = q;
}

__global__ void __launch_bounds__(256) k_bnreduce(const float* __restrict__ part,
                                                  float* __restrict__ sum,
                                                  float* __restrict__ sq) {
  const int d = threadIdx.x;
  float s = 0.f, q = 0.f;
  for (int b = 0; b < 256; ++b) {
    s += part[(size_t)b * 512 + d];
    q += part[(size_t)b * 512 + 256 + d];
  }
  sum[d] = s; sq[d] = q;
}

// ---- BN affine + IF node (scan over batch axis, per (t,d)) ---------------
__global__ void __launch_bounds__(256) k_if(const float* __restrict__ Y,
                                            const float* __restrict__ sum,
                                            const float* __restrict__ sq,
                                            const float* __restrict__ g,
                                            const float* __restrict__ be,
                                            h16* __restrict__ S) {
  const int idx = blockIdx.x * 256 + threadIdx.x;  // t*256 + d
  const int d = idx & 255, t = idx >> 8;
  const float inv = 1.0f / (float)M_;
  const float mu = sum[d] * inv;
  const float var = sq[d] * inv - mu * mu;
  const float sc = rsqrtf(var + 1e-5f) * g[d];
  const float sh = be[d];
  float v = 0.f;
  for (int b = 0; b < B_; ++b) {
    const size_t o = ((size_t)b * T_ + t) * DH_ + d;
    __builtin_prefetch((const void*)(Y + o + (size_t)T_ * DH_), 0, 0);
    v += (Y[o] - mu) * sc + sh;
    const float spk = (v >= 1.0f) ? 1.0f : 0.0f;   // VTH = 1, hard reset
    if (spk != 0.0f) v = 0.0f;
    S[o] = (h16)spk;
  }
}

// ---- boundary / reset logits: one wave per (b,t) row ---------------------
__global__ void __launch_bounds__(256) k_logits(const h16* __restrict__ H,
                                                const float* __restrict__ Wb,
                                                const float* __restrict__ bb,
                                                const float* __restrict__ Wr,
                                                const float* __restrict__ br,
                                                float* __restrict__ bl,
                                                float* __restrict__ rl) {
  const int lane = threadIdx.x & 31;
  const int row = blockIdx.x * 8 + (threadIdx.x >> 5);
  const h16* p = H + (size_t)row * DH_ + lane * 8;   // lane owns 8 channels
  v8h hv = *(const v8h*)p;
  float sb = 0.f, sr = 0.f;
#pragma unroll
  for (int e = 0; e < 8; ++e) {
    const float h = (float)hv[e];
    sb += h * Wb[lane * 8 + e];
    sr += h * Wr[lane * 8 + e];
  }
#pragma unroll
  for (int off = 16; off; off >>= 1) {
    sb += __shfl_xor(sb, off, 32);
    sr += __shfl_xor(sr, off, 32);
  }
  if (lane == 0) { bl[row] = sb + bb[0]; rl[row] = sr + br[0]; }
}

// ---- conditional LIF + boundaries + cumsum -> segments -------------------
__global__ void __launch_bounds__(256) k_lif(const float* __restrict__ bl,
                                             const float* __restrict__ rl,
                                             const int* __restrict__ mask,
                                             int* __restrict__ cnts,
                                             int* __restrict__ strt) {
  const int b = threadIdx.x;
  int* cnt = cnts + b * T_;
  for (int k = 0; k < T_; ++k) cnt[k] = 0;
  float v = 0.f, cum = 0.f;
  int prev = -1;
  for (int t = 0; t < T_; ++t) {
    const size_t o = (size_t)b * T_ + t;
    v += (bl[o] - v) * 0.5f;                  // tau = 2
    const float s = (v >= 1.0f) ? 1.0f : 0.0f;
    if (s != 0.0f) v = 0.0f;                  // hard reset
    if (rl[o] > 0.0f) v = 0.0f;               // sigmoid(x)>0.5 <=> x>0
    const float hard = s * (float)mask[o];
    const float bound = (t == 0) ? 1.0f : hard;
    cum += bound;
    int k = (int)(cum - 1.0f);
    k = k < 0 ? 0 : (k > T_ - 1 ? T_ - 1 : k);
    cnt[k]++;
    if (k != prev) { strt[b * T_ + k] = t; prev = k; }
  }
}

// ---- segment-mean of hidden spikes -> token_embs (f16) -------------------
__global__ void __launch_bounds__(256) k_tokemb(const h16* __restrict__ H,
                                                const int* __restrict__ cnts,
                                                const int* __restrict__ strt,
                                                h16* __restrict__ TE) {
  const int bk = blockIdx.x;        // b*512 + k
  const int d = threadIdx.x;
  const int b = bk >> 9;
  const int c = cnts[bk];
  float acc = 0.f;
  if (c > 0) {
    const int s = strt[bk];
    for (int t = s; t < s + c; ++t)
      acc += (float)H[((size_t)b * T_ + t) * DH_ + d];
    acc *= 1.0f / ((float)c + 1e-8f);
  }
  TE[(size_t)bk * DH_ + d] = (h16)acc;
}

// ---- GEMM3 (token_embs @ WpH + bp) fused with LayerNorm ------------------
__global__ void __launch_bounds__(256) k_gemm3ln(const h16* __restrict__ TE,
                                                 const h16* __restrict__ WpH,
                                                 const float* __restrict__ bp,
                                                 const float* __restrict__ lng,
                                                 const float* __restrict__ lnb,
                                                 float* __restrict__ out) {
  __shared__ float xs[16][DO_ + 8];
  __shared__ float mu_s[16], rs_s[16];
  const int lane = threadIdx.x & 31, wave = threadIdx.x >> 5;
  const int r = lane & 15, hi = lane >> 4;
  const int m0 = blockIdx.x * 16;
  const h16* arow = TE + (size_t)(m0 + r) * DH_;
  v8f acc[6];
#pragma unroll
  for (int j = 0; j < 6; ++j)
#pragma unroll
    for (int e = 0; e < 8; ++e) acc[j][e] = 0.0f;
#pragma unroll
  for (int k0 = 0; k0 < DH_; k0 += 32) {
    v16h a = load_a_h16(arow, k0, hi);
    const h16* brow = WpH + (size_t)k0 * DO_ + r;
#pragma unroll
    for (int j = 0; j < 6; ++j) {                 // wave owns 6 N-tiles (96 cols)
      const int n0 = (wave * 6 + j) * 16;
      v16h b;
#pragma unroll
      for (int e = 0; e < 16; ++e)
        b[e] = brow[(size_t)((e >> 3) * 16 + hi * 8 + (e & 7)) * DO_ + n0];
      acc[j] = wmma_f16(a, b, acc[j]);
    }
  }
  // stage 16x768 pre-LN tile in LDS (with bias)
#pragma unroll
  for (int j = 0; j < 6; ++j) {
    const int n = (wave * 6 + j) * 16 + r;
    const float bias = bp[n];
#pragma unroll
    for (int v = 0; v < 8; ++v) xs[v + hi * 8][n] = acc[j][v] + bias;
  }
  __syncthreads();
  if (threadIdx.x < 16) {
    const int row = threadIdx.x;
    float s = 0.f, q = 0.f;
    for (int n = 0; n < DO_; ++n) { const float x = xs[row][n]; s += x; q += x * x; }
    const float m = s * (1.0f / DO_);
    const float var = q * (1.0f / DO_) - m * m;
    mu_s[row] = m;
    rs_s[row] = rsqrtf(var + 1e-5f);
  }
  __syncthreads();
  for (int idx = threadIdx.x; idx < 16 * DO_; idx += 256) {
    const int row = idx / DO_, n = idx - row * DO_;
    out[(size_t)(m0 + row) * DO_ + n] =
        (xs[row][n] - mu_s[row]) * rs_s[row] * lng[n] + lnb[n];
  }
}

// ---- launch --------------------------------------------------------------
extern "C" void kernel_launch(void* const* d_in, const int* in_sizes, int n_in,
                              void* d_out, int out_size, void* d_ws, size_t ws_size,
                              hipStream_t stream) {
  (void)in_sizes; (void)n_in; (void)out_size; (void)ws_size;
  const int*   ids   = (const int*)d_in[0];
  const int*   bmask = (const int*)d_in[1];
  const float* emb   = (const float*)d_in[2];
  const float* W1    = (const float*)d_in[3];
  const float* b1    = (const float*)d_in[4];
  const float* g1    = (const float*)d_in[5];
  const float* be1   = (const float*)d_in[6];
  const float* W2    = (const float*)d_in[7];
  const float* b2    = (const float*)d_in[8];
  const float* g2    = (const float*)d_in[9];
  const float* be2   = (const float*)d_in[10];
  const float* Wb    = (const float*)d_in[11];
  const float* bb    = (const float*)d_in[12];
  const float* Wr    = (const float*)d_in[13];
  const float* br    = (const float*)d_in[14];
  const float* Wp    = (const float*)d_in[15];
  const float* bp    = (const float*)d_in[16];
  const float* lng   = (const float*)d_in[17];
  const float* lnb   = (const float*)d_in[18];

  char* w = (char*)d_ws;
  float* Y   = (float*)(w);                                  // 128 MB (Y1/Y2)
  h16*   S1  = (h16*)(w + 134217728);                        // 64 MB
  h16*   HID = (h16*)(w + 134217728 + 67108864);             // 64 MB
  size_t o = 268435456;
  float* stats = (float*)(w + o); o += 4096;                 // sum1|sq1|sum2|sq2
  float* part  = (float*)(w + o); o += 524288;
  float* bl    = (float*)(w + o); o += 524288;
  float* rl    = (float*)(w + o); o += 524288;
  int*   cnts  = (int*)(w + o);   o += 524288;
  int*   strt  = (int*)(w + o);   o += 524288;
  h16*   embH  = (h16*)(w + o);   o += 65536;   // 256*128
  h16*   W1H   = (h16*)(w + o);   o += 65536;   // 128*256
  h16*   W2H   = (h16*)(w + o);   o += 131072;  // 256*256
  h16*   WpH   = (h16*)(w + o);   o += 393216;  // 256*768
  h16*   TE    = (h16*)Y;   // reuse Y region once Y2 is consumed

  // one-time f32 -> f16 weight conversion (tiny)
  k_tohalf<<<32, 256, 0, stream>>>(emb, embH, (256 * DC_) / 4);
  k_tohalf<<<32, 256, 0, stream>>>(W1, W1H, (DC_ * DH_) / 4);
  k_tohalf<<<64, 256, 0, stream>>>(W2, W2H, (DH_ * DH_) / 4);
  k_tohalf<<<192, 256, 0, stream>>>(Wp, WpH, (DH_ * DO_) / 4);

  const dim3 gGemm(M_ / 16, 2, 1);
  k_gemm1<<<gGemm, 256, 0, stream>>>(ids, embH, W1H, b1, Y);
  k_bnstats<<<256, 256, 0, stream>>>(Y, part);
  k_bnreduce<<<1, 256, 0, stream>>>(part, stats, stats + 256);
  k_if<<<(T_ * DH_) / 256, 256, 0, stream>>>(Y, stats, stats + 256, g1, be1, S1);
  k_gemm2<<<gGemm, 256, 0, stream>>>(S1, W2H, b2, Y);
  k_bnstats<<<256, 256, 0, stream>>>(Y, part);
  k_bnreduce<<<1, 256, 0, stream>>>(part, stats + 512, stats + 768);
  k_if<<<(T_ * DH_) / 256, 256, 0, stream>>>(Y, stats + 512, stats + 768, g2, be2, HID);
  k_logits<<<M_ / 8, 256, 0, stream>>>(HID, Wb, bb, Wr, br, bl, rl);
  k_lif<<<1, 256, 0, stream>>>(bl, rl, bmask, cnts, strt);
  k_tokemb<<<M_, 256, 0, stream>>>(HID, cnts, strt, TE);
  k_gemm3ln<<<M_ / 16, 256, 0, stream>>>(TE, WpH, bp, lng, lnb, (float*)d_out);
}
